// CellPieceGraphTransformer_2241972928606
// MI455X (gfx1250) — compile-verified, matrix operand order checked by reading, not run on hardware
//
#include <hip/hip_runtime.h>
#include <hip/hip_bf16.h>
#include <math.h>

typedef __attribute__((ext_vector_type(16))) __bf16 v16bf;
typedef __attribute__((ext_vector_type(8)))  float  v8f;

#define N_CELLS  200000
#define N_PIECES 100000
#define N_EDGES  500000
#define N_GRAPHS 256
#define EMB   128
#define HEADS 4
#define HDIM  128
#define HD    512   // HEADS*HDIM
#define LAYERS 4
#define FC1   64
#define NEG_SLOPE 0.2f
#define NTILES (N_CELLS/16)       // 12500
#define BFRAGS_PER_LAYER 65536    // 4 kk * 32 ntile * 32 lane * 16 halves

// ---------------------------------------------------------------- utilities
__global__ void k_zero(float* p, int n) {
    for (int i = blockIdx.x * blockDim.x + threadIdx.x; i < n; i += gridDim.x * blockDim.x)
        p[i] = 0.f;
}

// count incoming edges per cell per piece-type (layer-invariant)
__global__ void k_counts(const int* __restrict__ px, const int* __restrict__ esrc,
                         const int* __restrict__ edst, float* c0, float* c1) {
    for (int e = blockIdx.x * blockDim.x + threadIdx.x; e < N_EDGES; e += gridDim.x * blockDim.x) {
        int t = px[esrc[e]];
        atomicAdd((t ? c1 : c0) + edst[e], 1.0f);
    }
}

// S[l][t][512] = piece_emb[t] @ W_l[l] + b_l[l]  (only 2 distinct source rows!)
__global__ void k_stables(const float* __restrict__ piece_emb, const float* __restrict__ W_l,
                          const float* __restrict__ b_l, float* Stab) {
    int idx = blockIdx.x * blockDim.x + threadIdx.x;
    if (idx >= LAYERS * 2 * HD) return;
    int l = idx >> 10, t = (idx >> 9) & 1, n = idx & 511;
    float s = b_l[l * HD + n];
    for (int k = 0; k < EMB; ++k)
        s += piece_emb[t * EMB + k] * W_l[(l * EMB + k) * HD + n];
    Stab[idx] = s;
}

// Swizzle W_r (fp32 [L][128][512]) into exact bf16 WMMA B-fragment order:
// flat = ((kk*32 + ntile)*32 + lane)*16 + i ;  K = kk*32 + (i>>3)*16 + (lane>>4)*8 + (i&7)
// N = ntile*16 + (lane&15)
__global__ void k_wswz(const float* __restrict__ Wr, __bf16* Bsw) {
    int idx = blockIdx.x * blockDim.x + threadIdx.x;
    if (idx >= LAYERS * BFRAGS_PER_LAYER) return;
    int l = idx >> 16, f = idx & 65535;
    int i = f & 15, lane = (f >> 4) & 31, nt = (f >> 9) & 31, kk = f >> 14;
    int kl = ((i >> 3) << 4) + ((lane >> 4) << 3) + (i & 7);
    int k = kk * 32 + kl;
    int n = (nt << 4) + (lane & 15);
    Bsw[idx] = (__bf16)Wr[((l * EMB) + k) * HD + n];
}

// initial cell state = cell_emb[cell_x]
__global__ void k_init(const int* __restrict__ cell_x, const float* __restrict__ cell_emb,
                       float* cell) {
    int total = N_CELLS * EMB / 4;
    for (int i4 = blockIdx.x * blockDim.x + threadIdx.x; i4 < total; i4 += gridDim.x * blockDim.x) {
        int row = i4 >> 5, c4 = i4 & 31;
        int ce = cell_x[row];
        ((float4*)cell)[i4] = ((const float4*)(cell_emb + (size_t)ce * EMB))[c4];
    }
}

// ---------------------------------------------------------------- fused GATv2 layer
// Per 16-cell tile (one wave): x_dst tile = cell_tile @ W_r + b_r via WMMA (kept in regs),
// logits vs the 2 source rows, count-weighted 2-term softmax, head-mean recombine, ReLU.
__global__ __launch_bounds__(256)
void k_layer(const float* __restrict__ cell_in, float* __restrict__ cell_out,
             const float* __restrict__ Stab_l,   // [2][512]
             const float* __restrict__ att_l,    // [4][128]
             const float* __restrict__ br_l,     // [512]
             const float* __restrict__ bias_l,   // [128]
             const float* __restrict__ c0, const float* __restrict__ c1,
             const __bf16* __restrict__ Bsw_l)   // [65536] fragment-ordered
{
    __shared__ float S0s[HD], S1s[HD], atts[HD], brs[HD], biass[EMB];
    __shared__ float coefs[8 * 128];   // 8 waves x [16 rows][4 heads][2 types]

    for (int i = threadIdx.x; i < HD; i += blockDim.x) {
        S0s[i] = Stab_l[i];
        S1s[i] = Stab_l[HD + i];
        atts[i] = att_l[i];
        brs[i] = br_l[i];
    }
    for (int i = threadIdx.x; i < EMB; i += blockDim.x) biass[i] = bias_l[i];
    __syncthreads();

    const int wave = threadIdx.x >> 5;
    const int lane = threadIdx.x & 31;
    const int m = lane & 15;      // row within tile (A) / column within N-tile (B,D)
    const int hi = lane >> 4;     // half-wave select
    float* mycoef = coefs + wave * 128;

    const int gwave = blockIdx.x * (blockDim.x >> 5) + wave;
    const int nwaves = gridDim.x * (blockDim.x >> 5);
    const v16bf* Bv = (const v16bf*)Bsw_l;

    for (int tile = gwave; tile < NTILES; tile += nwaves) {
        const int row0 = tile * 16;

        // ---- A fragments (16x32 bf16, per documented per-lane layout), 4 k-steps
        v16bf afrag[4];
        const float* rowp = cell_in + (size_t)(row0 + m) * EMB;
        #pragma unroll
        for (int kk = 0; kk < 4; ++kk) {
            #pragma unroll
            for (int g = 0; g < 2; ++g) {
                const float4* p = (const float4*)(rowp + kk * 32 + g * 16 + hi * 8);
                float4 f0 = p[0], f1 = p[1];
                afrag[kk][g * 8 + 0] = (__bf16)f0.x;
                afrag[kk][g * 8 + 1] = (__bf16)f0.y;
                afrag[kk][g * 8 + 2] = (__bf16)f0.z;
                afrag[kk][g * 8 + 3] = (__bf16)f0.w;
                afrag[kk][g * 8 + 4] = (__bf16)f1.x;
                afrag[kk][g * 8 + 5] = (__bf16)f1.y;
                afrag[kk][g * 8 + 6] = (__bf16)f1.z;
                afrag[kk][g * 8 + 7] = (__bf16)f1.w;
            }
        }

        // edge-type counts for the row this lane owns (lanes m<8 own row hi*8+m)
        float oc0 = 0.f, oc1 = 0.f;
        if (m < 8) {
            int orow = row0 + hi * 8 + m;
            oc0 = c0[orow];
            oc1 = c1[orow];
        }

        // ---- per head: GEMM (8 N-tiles x 4 k-steps of WMMA) + logit reduction
        #pragma unroll 1
        for (int h = 0; h < HEADS; ++h) {
            v8f z8 = {0.f, 0.f, 0.f, 0.f, 0.f, 0.f, 0.f, 0.f};
            v8f acc[8] = {z8, z8, z8, z8, z8, z8, z8, z8};
            #pragma unroll
            for (int nn = 0; nn < 8; ++nn) {
                const int nt = h * 8 + nn;
                #pragma unroll
                for (int kk = 0; kk < 4; ++kk) {
                    v16bf bfr = Bv[(kk * 32 + nt) * 32 + lane];
                    acc[nn] = __builtin_amdgcn_wmma_f32_16x16x32_bf16(
                        false, afrag[kk], false, bfr, (short)0, acc[nn], false, false);
                }
            }
            // logit partials: sum_n a[n]*lrelu(S_t[n] + xdst[m][n])  (D layout: row=hi*8+v)
            float p0[8] = {0, 0, 0, 0, 0, 0, 0, 0};
            float p1[8] = {0, 0, 0, 0, 0, 0, 0, 0};
            #pragma unroll
            for (int nn = 0; nn < 8; ++nn) {
                const int d = nn * 16 + m;
                const float s0v = S0s[h * HDIM + d];
                const float s1v = S1s[h * HDIM + d];
                const float av  = atts[h * HDIM + d];
                const float brv = brs[h * HDIM + d];
                #pragma unroll
                for (int v = 0; v < 8; ++v) {
                    float x = acc[nn][v] + brv;
                    float z0 = s0v + x; z0 = (z0 > 0.f) ? z0 : NEG_SLOPE * z0;
                    float z1 = s1v + x; z1 = (z1 > 0.f) ? z1 : NEG_SLOPE * z1;
                    p0[v] += av * z0;
                    p1[v] += av * z1;
                }
            }
            // reduce across the 16 lanes of each half-wave
            #pragma unroll
            for (int v = 0; v < 8; ++v) {
                #pragma unroll
                for (int msk = 1; msk < 16; msk <<= 1) {
                    p0[v] += __shfl_xor(p0[v], msk, 32);
                    p1[v] += __shfl_xor(p1[v], msk, 32);
                }
            }
            // count-weighted 2-term segment softmax -> per-row head coefficients
            #pragma unroll
            for (int v = 0; v < 8; ++v) {
                if (m == v) {
                    float L0 = p0[v], L1 = p1[v];
                    float A0 = 0.f, A1 = 0.f;
                    if (oc0 > 0.f || oc1 > 0.f) {
                        float mx = (oc0 > 0.f && oc1 > 0.f) ? fmaxf(L0, L1)
                                                            : (oc0 > 0.f ? L0 : L1);
                        float e0 = (oc0 > 0.f) ? oc0 * __expf(L0 - mx) : 0.f;
                        float e1 = (oc1 > 0.f) ? oc1 * __expf(L1 - mx) : 0.f;
                        float den = e0 + e1;
                        A0 = e0 / den;
                        A1 = e1 / den;
                    }
                    int r = hi * 8 + v;
                    mycoef[(r * 4 + h) * 2 + 0] = A0;
                    mycoef[(r * 4 + h) * 2 + 1] = A1;
                }
            }
        }

        __builtin_amdgcn_wave_barrier();
        asm volatile("s_wait_dscnt 0" ::: "memory");

        // ---- recombine: out[r,:] = relu(mean_h(A0*S0[h]+A1*S1[h]) + bias)
        {
            const int r = lane >> 1;
            const int cbase = (lane & 1) * 64;
            float Ah[HEADS][2];
            #pragma unroll
            for (int h = 0; h < HEADS; ++h) {
                Ah[h][0] = mycoef[(r * 4 + h) * 2 + 0];
                Ah[h][1] = mycoef[(r * 4 + h) * 2 + 1];
            }
            float* op = cell_out + (size_t)(row0 + r) * EMB + cbase;
            #pragma unroll
            for (int c = 0; c < 64; c += 4) {
                float tmp[4];
                #pragma unroll
                for (int j = 0; j < 4; ++j) {
                    const int d = cbase + c + j;
                    float s = biass[d];
                    #pragma unroll
                    for (int h = 0; h < HEADS; ++h)
                        s += 0.25f * (Ah[h][0] * S0s[h * HDIM + d] +
                                      Ah[h][1] * S1s[h * HDIM + d]);
                    tmp[j] = fmaxf(s, 0.f);
                }
                float4 o; o.x = tmp[0]; o.y = tmp[1]; o.z = tmp[2]; o.w = tmp[3];
                *(float4*)(op + c) = o;
            }
        }
    }
}

// ---------------------------------------------------------------- pooling (sorted batch)
__global__ void k_pool(const float* __restrict__ cell, const int* __restrict__ batch,
                       float* gsum, float* gcnt) {
    const int start = blockIdx.x * 64;       // 3125 blocks x 64 cells
    const int col = threadIdx.x;             // 128 threads = one column each
    int g = batch[start];
    float acc = 0.f, cnt = 0.f;
    for (int i = 0; i < 64; ++i) {
        int row = start + i;
        int gb = batch[row];
        if (gb != g) {
            atomicAdd(&gsum[(size_t)g * EMB + col], acc);
            if (col == 0) atomicAdd(&gcnt[g], cnt);
            acc = 0.f; cnt = 0.f; g = gb;
        }
        acc += cell[(size_t)row * EMB + col];
        cnt += 1.f;
    }
    atomicAdd(&gsum[(size_t)g * EMB + col], acc);
    if (col == 0) atomicAdd(&gcnt[g], cnt);
}

// ---------------------------------------------------------------- MLP heads
__global__ void k_head(const float* __restrict__ gsum, const float* __restrict__ gcnt,
                       const float* __restrict__ fc1_w, const float* __restrict__ fc1_b,
                       const float* __restrict__ pw, const float* __restrict__ pb,
                       const float* __restrict__ vw, const float* __restrict__ vb,
                       float* out) {
    int g = blockIdx.x * blockDim.x + threadIdx.x;
    if (g >= N_GRAPHS) return;
    float inv = 1.0f / fmaxf(gcnt[g], 1.0f);
    float h[FC1];
    for (int j = 0; j < FC1; ++j) {
        float s = fc1_b[j];
        for (int k = 0; k < EMB; ++k)
            s += gsum[(size_t)g * EMB + k] * inv * fc1_w[k * FC1 + j];
        h[j] = fmaxf(s, 0.f);
    }
    for (int o = 0; o < 8; ++o) {
        float s = pb[o];
        for (int j = 0; j < FC1; ++j) s += h[j] * pw[j * 8 + o];
        out[g * 8 + o] = s;
    }
    float v = vb[0];
    for (int j = 0; j < FC1; ++j) v += h[j] * vw[j];
    out[N_GRAPHS * 8 + g] = tanhf(v);
}

// ---------------------------------------------------------------- launch
extern "C" void kernel_launch(void* const* d_in, const int* in_sizes, int n_in,
                              void* d_out, int out_size, void* d_ws, size_t ws_size,
                              hipStream_t stream) {
    const int*   cell_x    = (const int*)d_in[0];
    const int*   piece_x   = (const int*)d_in[1];
    const int*   esrc      = (const int*)d_in[2];
    const int*   edst      = (const int*)d_in[3];
    const int*   cbatch    = (const int*)d_in[4];
    const float* cell_emb  = (const float*)d_in[5];
    const float* piece_emb = (const float*)d_in[6];
    const float* W_l       = (const float*)d_in[7];
    const float* b_l       = (const float*)d_in[8];
    const float* W_r       = (const float*)d_in[9];
    const float* b_r       = (const float*)d_in[10];
    const float* att       = (const float*)d_in[11];
    const float* conv_bias = (const float*)d_in[12];
    const float* fc1_w     = (const float*)d_in[13];
    const float* fc1_b     = (const float*)d_in[14];
    const float* policy_w  = (const float*)d_in[15];
    const float* policy_b  = (const float*)d_in[16];
    const float* value_w   = (const float*)d_in[17];
    const float* value_b   = (const float*)d_in[18];

    char* ws = (char*)d_ws;
    float*  cellA = (float*)(ws);                       // 102,400,000 B
    float*  cellB = (float*)(ws + 102400000);           // 102,400,000 B
    float*  c0    = (float*)(ws + 204800000);           //     800,000 B
    float*  c1    = (float*)(ws + 205600000);           //     800,000 B
    float*  gsum  = (float*)(ws + 206400000);           //     131,072 B
    float*  gcnt  = (float*)(ws + 206531072);           //       1,024 B
    float*  Stab  = (float*)(ws + 206532096);           //      16,384 B
    __bf16* Bsw   = (__bf16*)(ws + 206548480);          //     524,288 B

    // zero c0,c1,gsum,gcnt (contiguous): 433,024 floats
    k_zero<<<512, 256, 0, stream>>>(c0, 433024);
    k_counts<<<512, 256, 0, stream>>>(piece_x, esrc, edst, c0, c1);
    k_stables<<<16, 256, 0, stream>>>(piece_emb, W_l, b_l, Stab);
    k_wswz<<<1024, 256, 0, stream>>>(W_r, Bsw);
    k_init<<<2048, 256, 0, stream>>>(cell_x, cell_emb, cellA);

    float* bufs[2] = {cellA, cellB};
    for (int l = 0; l < LAYERS; ++l) {
        k_layer<<<192, 256, 0, stream>>>(bufs[l & 1], bufs[(l + 1) & 1],
                                         Stab + l * 1024, att + l * 512,
                                         b_r + l * 512, conv_bias + l * 128,
                                         c0, c1, Bsw + l * BFRAGS_PER_LAYER);
    }
    // after 4 layers final state is in cellA
    k_pool<<<3125, 128, 0, stream>>>(cellA, cbatch, gsum, gcnt);
    k_head<<<1, 256, 0, stream>>>(gsum, gcnt, fc1_w, fc1_b,
                                  policy_w, policy_b, value_w, value_b, (float*)d_out);
}
